// ALSHVGGNet_7198365188562
// MI455X (gfx1250) — matrix operand order repaired and checked by reading, not compiled, not run on hardware
//
#include <hip/hip_runtime.h>
#include <hip/hip_bf16.h>

typedef __attribute__((ext_vector_type(8)))  _Float16 v8h;
typedef __attribute__((ext_vector_type(16))) _Float16 v16h;
typedef __attribute__((ext_vector_type(4)))  float    v4f;
typedef __attribute__((ext_vector_type(8)))  float    v8f;

#define EPSF 1e-5f
#define UCONST 0.999f
#define LDA 40   // LDS row stride in halves (80B): 16B-aligned, bank-conflict-friendly

static __device__ __forceinline__ v16h cat8(v8h lo, v8h hi) {
    return __builtin_shufflevector(lo, hi, 0, 1, 2, 3, 4, 5, 6, 7,
                                           8, 9, 10, 11, 12, 13, 14, 15);
}

// ---------------------------------------------------------------------------
// Implicit-GEMM 3x3 conv (stride 1, pad 1, NCHW/OIHW) via WMMA f16->f32.
// Block = 128 threads (4 wave32s). Tile: M=32 output channels (2 accumulators
// per wave), N=64 pixels (16 per wave), K = Cin*9 in chunks of 32 staged in
// LDS as f16.  A: [m][k] rows, B: [p][k] rows -> all fragment reads are
// contiguous 16B runs (ds_load_b128).  Bias + optional ALSH mask fused.
// ---------------------------------------------------------------------------
__global__ __launch_bounds__(128) void conv3x3_wmma_kernel(
    const float* __restrict__ in, const float* __restrict__ wgt,
    const float* __restrict__ bias, const float* __restrict__ mask,
    float* __restrict__ out, int Cin, int Cout, int H, int W, int B)
{
    const int K    = Cin * 9;
    const int HW   = H * W;
    const int Npix = B * HW;
    const int pixBase  = blockIdx.x * 64;
    const int coutBase = blockIdx.y * 32;
    const int tid  = threadIdx.x;
    const int lane = tid & 31;
    const int wv   = tid >> 5;

    __shared__ __align__(16) _Float16 As[32 * LDA];  // [m][k]
    __shared__ __align__(16) _Float16 Bs[64 * LDA];  // [p][k]

    v8f acc0 = {}, acc1 = {};

    // Fixed per-thread staging coordinates.
    const int rA  = tid >> 2;          // A row 0..31
    const int k0A = (tid & 3) << 3;    // A k-offset 0,8,16,24
    const int pB  = tid & 63;          // B pixel slot 0..63
    const int khB = (tid >> 6) << 4;   // B k-offset 0 or 16
    const int pixB = pixBase + pB;
    const bool pixOK = pixB < Npix;
    int bB = 0, oyB = 0, oxB = 0;
    if (pixOK) {
        bB = pixB / HW;
        int s = pixB - bB * HW;
        oyB = s / W;
        oxB = s - oyB * W;
    }
    const bool alignedK = ((K & 7) == 0);
    const float* __restrict__ wrowA = wgt + (long)(coutBase + rA) * K;

    for (int kt = 0; kt < K; kt += 32) {
        // -------- stage A: 32x32 weights, 8 contiguous k per thread --------
        {
            _Float16 tmp[8];
            if (alignedK && (kt + 32 <= K)) {
                v4f a0 = *(const v4f*)(wrowA + kt + k0A);
                v4f a1 = *(const v4f*)(wrowA + kt + k0A + 4);
                #pragma unroll
                for (int i = 0; i < 4; ++i) {
                    tmp[i]     = (_Float16)a0[i];
                    tmp[4 + i] = (_Float16)a1[i];
                }
            } else {
                #pragma unroll
                for (int i = 0; i < 8; ++i) {
                    int k = kt + k0A + i;
                    tmp[i] = (k < K) ? (_Float16)wrowA[k] : (_Float16)0.0f;
                }
            }
            *(v8h*)&As[rA * LDA + k0A] = *(v8h*)tmp;
        }
        if (kt + 32 < K)
            __builtin_prefetch(wrowA + kt + 32 + k0A, 0, 0);

        // -------- stage B: im2col, 16 contiguous k per thread --------------
        {
            _Float16 tmp[16];
            int k  = kt + khB;
            int ci = k / 9;
            int r9 = k - ci * 9;
            #pragma unroll
            for (int i = 0; i < 16; ++i, ++k) {
                float v = 0.0f;
                if (pixOK && k < K) {
                    int ky = r9 / 3;
                    int kx = r9 - ky * 3;
                    int iy = oyB + ky - 1;
                    int ix = oxB + kx - 1;
                    if (iy >= 0 && iy < H && ix >= 0 && ix < W)
                        v = in[((long)(bB * Cin + ci) * H + iy) * W + ix];
                }
                tmp[i] = (_Float16)v;
                if (++r9 == 9) { r9 = 0; ++ci; }
            }
            *(v8h*)&Bs[pB * LDA + khB]     = *(v8h*)&tmp[0];
            *(v8h*)&Bs[pB * LDA + khB + 8] = *(v8h*)&tmp[8];
        }
        __syncthreads();

        // -------- fragments: contiguous 16B LDS loads ----------------------
        const int ml  = lane & 15;
        const int kbA = (lane >= 16) ? 16 : 0;
        const int kbB = (lane >= 16) ? 16 : 0;
        v16h af0 = cat8(*(const v8h*)&As[ml * LDA + kbA],
                        *(const v8h*)&As[ml * LDA + kbA + 8]);
        v16h af1 = cat8(*(const v8h*)&As[(ml + 16) * LDA + kbA],
                        *(const v8h*)&As[(ml + 16) * LDA + kbA + 8]);
        v16h bf  = *(const v16h*)&Bs[(wv * 16 + ml) * LDA + kbB];

        acc0 = __builtin_amdgcn_wmma_f32_16x16x32_f16(
                   false, af0, false, bf, (short)0, acc0, false, false);
        acc1 = __builtin_amdgcn_wmma_f32_16x16x32_f16(
                   false, af1, false, bf, (short)0, acc1, false, false);
        __syncthreads();
    }

    // -------- store: C layout lane l, vgpr r -> M = r + (l>=16?8:0), N=l&15
    {
        const int nl  = lane & 15;
        const int pix = pixBase + wv * 16 + nl;
        if (pix < Npix) {
            int b  = pix / HW;
            int s  = pix - b * HW;
            int oy = s / W;
            int ox = s - oy * W;
            long base = ((long)b * Cout) * HW + (long)oy * W + ox;
            #pragma unroll
            for (int r = 0; r < 8; ++r) {
                int mrow = r + ((lane >= 16) ? 8 : 0);
                int co0  = coutBase + mrow;
                int co1  = coutBase + 16 + mrow;
                float v0 = acc0[r] + bias[co0];
                float v1 = acc1[r] + bias[co1];
                if (mask) { v0 *= mask[co0]; v1 *= mask[co1]; }
                out[base + (long)co0 * HW] = v0;
                out[base + (long)co1 * HW] = v1;
            }
        }
    }
}

// ---------------------------------------------------------------------------
// WMMA GEMM: out[M,N] = A[M,K] * Wt[N,K]^T + bias[N].
// Tile: M=32 (2 accumulators/wave), N=64 (16 per wave), K chunks of 32.
// ---------------------------------------------------------------------------
__global__ __launch_bounds__(128) void gemm_wmma_kernel(
    const float* __restrict__ A, const float* __restrict__ Wt,
    const float* __restrict__ bias, float* __restrict__ out,
    int M, int N, int K)
{
    const int mBase = blockIdx.y * 32;
    const int nBase = blockIdx.x * 64;
    const int tid  = threadIdx.x;
    const int lane = tid & 31;
    const int wv   = tid >> 5;

    __shared__ __align__(16) _Float16 As[32 * LDA];
    __shared__ __align__(16) _Float16 Bs[64 * LDA];

    v8f acc0 = {}, acc1 = {};

    const int rA  = tid >> 2;
    const int k0A = (tid & 3) << 3;
    const int pB  = tid & 63;
    const int khB = (tid >> 6) << 4;
    const int mA  = mBase + rA;
    const int nB  = nBase + pB;
    const bool alignedK = ((K & 7) == 0);

    for (int kt = 0; kt < K; kt += 32) {
        // -------- stage A --------
        {
            _Float16 tmp[8];
            if (mA < M && alignedK && (kt + 32 <= K)) {
                const float* src = A + (long)mA * K + kt + k0A;
                v4f a0 = *(const v4f*)src;
                v4f a1 = *(const v4f*)(src + 4);
                #pragma unroll
                for (int i = 0; i < 4; ++i) {
                    tmp[i]     = (_Float16)a0[i];
                    tmp[4 + i] = (_Float16)a1[i];
                }
            } else {
                #pragma unroll
                for (int i = 0; i < 8; ++i) {
                    int k = kt + k0A + i;
                    tmp[i] = (mA < M && k < K) ? (_Float16)A[(long)mA * K + k]
                                               : (_Float16)0.0f;
                }
            }
            *(v8h*)&As[rA * LDA + k0A] = *(v8h*)tmp;
        }
        // -------- stage B (weights, contiguous in k) --------
        {
            _Float16 tmp[16];
            if (nB < N && alignedK && (kt + 32 <= K)) {
                const float* src = Wt + (long)nB * K + kt + khB;
                #pragma unroll
                for (int q = 0; q < 4; ++q) {
                    v4f b4 = *(const v4f*)(src + q * 4);
                    #pragma unroll
                    for (int i = 0; i < 4; ++i)
                        tmp[q * 4 + i] = (_Float16)b4[i];
                }
                if (kt + 64 <= K) __builtin_prefetch(src + 32, 0, 0);
            } else {
                #pragma unroll
                for (int i = 0; i < 16; ++i) {
                    int k = kt + khB + i;
                    tmp[i] = (nB < N && k < K) ? (_Float16)Wt[(long)nB * K + k]
                                               : (_Float16)0.0f;
                }
            }
            *(v8h*)&Bs[pB * LDA + khB]     = *(v8h*)&tmp[0];
            *(v8h*)&Bs[pB * LDA + khB + 8] = *(v8h*)&tmp[8];
        }
        __syncthreads();

        const int ml  = lane & 15;
        const int kbA = (lane >= 16) ? 16 : 0;
        const int kbB = (lane >= 16) ? 16 : 0;
        v16h af0 = cat8(*(const v8h*)&As[ml * LDA + kbA],
                        *(const v8h*)&As[ml * LDA + kbA + 8]);
        v16h af1 = cat8(*(const v8h*)&As[(ml + 16) * LDA + kbA],
                        *(const v8h*)&As[(ml + 16) * LDA + kbA + 8]);
        v16h bf  = *(const v16h*)&Bs[(wv * 16 + ml) * LDA + kbB];

        acc0 = __builtin_amdgcn_wmma_f32_16x16x32_f16(
                   false, af0, false, bf, (short)0, acc0, false, false);
        acc1 = __builtin_amdgcn_wmma_f32_16x16x32_f16(
                   false, af1, false, bf, (short)0, acc1, false, false);
        __syncthreads();
    }

    {
        const int nl  = lane & 15;
        const int col = nBase + wv * 16 + nl;
        if (col < N) {
            float bv = bias[col];
            #pragma unroll
            for (int r = 0; r < 8; ++r) {
                int mrow = r + ((lane >= 16) ? 8 : 0);
                int row0 = mBase + mrow;
                int row1 = mBase + 16 + mrow;
                if (row0 < M) out[(long)row0 * N + col] = acc0[r] + bv;
                if (row1 < M) out[(long)row1 * N + col] = acc1[r] + bv;
            }
        }
    }
}

// ---------------------------------------------------------------------------
// BN2d batch statistics: one block per channel, mean + rsqrt(var+eps).
// ---------------------------------------------------------------------------
__global__ __launch_bounds__(256) void bn2d_stats_kernel(
    const float* __restrict__ x, float* __restrict__ mean, float* __restrict__ rstd,
    int C, int HW, int B)
{
    const int c   = blockIdx.x;
    const int tid = threadIdx.x;
    const int Nt  = B * HW;
    float s = 0.f, s2 = 0.f;
    for (int t = tid; t < Nt; t += 256) {
        int b = t / HW;
        int i = t - b * HW;
        float v = x[((long)(b * C + c)) * HW + i];
        s += v; s2 += v * v;
    }
    __shared__ float sh[256], sh2[256];
    sh[tid] = s; sh2[tid] = s2;
    __syncthreads();
    for (int o = 128; o > 0; o >>= 1) {
        if (tid < o) { sh[tid] += sh[tid + o]; sh2[tid] += sh2[tid + o]; }
        __syncthreads();
    }
    if (tid == 0) {
        float m = sh[0] / (float)Nt;
        float v = sh2[0] / (float)Nt - m * m;
        mean[c] = m;
        rstd[c] = rsqrtf(v + EPSF);
    }
}

__global__ __launch_bounds__(256) void bn2d_apply_relu_kernel(
    float* __restrict__ x, const float* __restrict__ mean, const float* __restrict__ rstd,
    const float* __restrict__ g, const float* __restrict__ be,
    int C, int HW, long total)
{
    long i = (long)blockIdx.x * 256 + threadIdx.x;
    if (i >= total) return;
    int c = (int)((i / HW) % C);
    float v = (x[i] - mean[c]) * rstd[c] * g[c] + be[c];
    x[i] = v > 0.f ? v : 0.f;
}

__global__ __launch_bounds__(256) void maxpool2_kernel(
    const float* __restrict__ in, float* __restrict__ out,
    int C, int H, int W, int B)
{
    const int Ho = H >> 1, Wo = W >> 1;
    long total = (long)B * C * Ho * Wo;
    long i = (long)blockIdx.x * 256 + threadIdx.x;
    if (i >= total) return;
    int ox = (int)(i % Wo); long t = i / Wo;
    int oy = (int)(t % Ho); t /= Ho;
    int c  = (int)(t % C);
    int b  = (int)(t / C);
    const float* p = in + (((long)(b * C + c) * H + oy * 2) * W + ox * 2);
    float v = fmaxf(fmaxf(p[0], p[1]), fmaxf(p[W], p[W + 1]));
    out[i] = v;
}

__global__ __launch_bounds__(256) void bn1d_stats_kernel(
    const float* __restrict__ x, float* __restrict__ mean, float* __restrict__ rstd,
    int M, int F)
{
    const int f   = blockIdx.x;
    const int tid = threadIdx.x;
    float s = 0.f, s2 = 0.f;
    for (int m = tid; m < M; m += 256) {
        float v = x[(long)m * F + f];
        s += v; s2 += v * v;
    }
    __shared__ float sh[256], sh2[256];
    sh[tid] = s; sh2[tid] = s2;
    __syncthreads();
    for (int o = 128; o > 0; o >>= 1) {
        if (tid < o) { sh[tid] += sh[tid + o]; sh2[tid] += sh2[tid + o]; }
        __syncthreads();
    }
    if (tid == 0) {
        float m = sh[0] / (float)M;
        float v = sh2[0] / (float)M - m * m;
        mean[f] = m;
        rstd[f] = rsqrtf(v + EPSF);
    }
}

__global__ __launch_bounds__(256) void bn1d_apply_relu_kernel(
    float* __restrict__ x, const float* __restrict__ mean, const float* __restrict__ rstd,
    const float* __restrict__ g, const float* __restrict__ be, int F, long total)
{
    long i = (long)blockIdx.x * 256 + threadIdx.x;
    if (i >= total) return;
    int f = (int)(i % F);
    float v = (x[i] - mean[f]) * rstd[f] * g[f] + be[f];
    x[i] = v > 0.f ? v : 0.f;
}

// ---------------------------------------------------------------------------
// ALSH pieces
// ---------------------------------------------------------------------------
__global__ __launch_bounds__(256) void filter_norms_kernel(
    const float* __restrict__ w, float* __restrict__ norms, int K)
{
    const int o   = blockIdx.x;
    const int tid = threadIdx.x;
    float s = 0.f;
    for (int k = tid; k < K; k += 256) {
        float v = w[(long)o * K + k];
        s += v * v;
    }
    __shared__ float sh[256];
    sh[tid] = s;
    __syncthreads();
    for (int off = 128; off > 0; off >>= 1) {
        if (tid < off) sh[tid] += sh[tid + off];
        __syncthreads();
    }
    if (tid == 0) norms[o] = sqrtf(sh[0]);
}

// One block, 256 threads (one per filter).  hash_a: (2, 2306) row-major.
__global__ __launch_bounds__(256) void alsh_mask_kernel(
    const float* __restrict__ w6, const float* __restrict__ hash_a,
    const float* __restrict__ norms, const float* __restrict__ q,
    float* __restrict__ mask, int K /*2304*/, int Cq /*256*/)
{
    const int tid = threadIdx.x;
    const int HD  = K + 2;  // 2306
    __shared__ float sh[256];
    __shared__ float dq[2];
    __shared__ int   codeq;

    // max filter norm
    sh[tid] = norms[tid];
    __syncthreads();
    for (int o = 128; o > 0; o >>= 1) {
        if (tid < o) sh[tid] = fmaxf(sh[tid], sh[tid + o]);
        __syncthreads();
    }
    const float scale = UCONST / sh[0];
    __syncthreads();

    // ||q tiled 9x||
    float qc = q[tid];
    sh[tid] = qc * qc;
    __syncthreads();
    for (int o = 128; o > 0; o >>= 1) {
        if (tid < o) sh[tid] += sh[tid + o];
        __syncthreads();
    }
    const float qnorm = sqrtf(9.f * sh[0]) + 1e-12f;
    __syncthreads();

    // query dots against both hash rows (cooperative over K)
    float d0 = 0.f, d1 = 0.f;
    for (int k = tid; k < K; k += 256) {
        float qk = q[k % Cq] / qnorm;
        d0 += qk * hash_a[k];
        d1 += qk * hash_a[HD + k];
    }
    sh[tid] = d0;
    __syncthreads();
    for (int o = 128; o > 0; o >>= 1) {
        if (tid < o) sh[tid] += sh[tid + o];
        __syncthreads();
    }
    if (tid == 0) dq[0] = sh[0];
    __syncthreads();
    sh[tid] = d1;
    __syncthreads();
    for (int o = 128; o > 0; o >>= 1) {
        if (tid < o) sh[tid] += sh[tid + o];
        __syncthreads();
    }
    if (tid == 0) {
        dq[1] = sh[0];
        codeq = (dq[0] > 0.f ? 1 : 0) + (dq[1] > 0.f ? 2 : 0);
    }
    __syncthreads();

    // per-filter code
    const float ns  = norms[tid] * scale;
    const float ns2 = ns * ns;
    const float ns4 = ns2 * ns2;
    float f0 = ns2 * hash_a[K] + ns4 * hash_a[K + 1];
    float f1 = ns2 * hash_a[HD + K] + ns4 * hash_a[HD + K + 1];
    const float* wr = w6 + (long)tid * K;
    for (int k = 0; k < K; ++k) {
        float wv = wr[k] * scale;
        f0 += wv * hash_a[k];
        f1 += wv * hash_a[HD + k];
    }
    int cf = (f0 > 0.f ? 1 : 0) + (f1 > 0.f ? 2 : 0);
    mask[tid] = (cf == codeq) ? 1.f : 0.f;
}

// ---------------------------------------------------------------------------
// Host orchestration
// ---------------------------------------------------------------------------
extern "C" void kernel_launch(void* const* d_in, const int* in_sizes, int n_in,
                              void* d_out, int out_size, void* d_ws, size_t ws_size,
                              hipStream_t stream)
{
    (void)in_sizes; (void)n_in; (void)out_size; (void)ws_size;

    const float* x = (const float*)d_in[0];
    const float *w[7], *bb[7], *g[7], *be[7];
    for (int i = 1; i <= 6; ++i) {
        w[i]  = (const float*)d_in[(i - 1) * 4 + 1];
        bb[i] = (const float*)d_in[(i - 1) * 4 + 2];
        g[i]  = (const float*)d_in[(i - 1) * 4 + 3];
        be[i] = (const float*)d_in[(i - 1) * 4 + 4];
    }
    const float* hash_a = (const float*)d_in[25];
    const float* fc7_w  = (const float*)d_in[26];
    const float* fc7_b  = (const float*)d_in[27];
    const float* g7     = (const float*)d_in[28];
    const float* be7    = (const float*)d_in[29];
    const float* fc8_w  = (const float*)d_in[30];
    const float* fc8_b  = (const float*)d_in[31];
    const float* g8     = (const float*)d_in[32];
    const float* be8    = (const float*)d_in[33];
    const float* fc9_w  = (const float*)d_in[34];
    const float* fc9_b  = (const float*)d_in[35];
    float* out = (float*)d_out;

    // Workspace layout: two 64 MiB ping-pong activation buffers + stats.
    float* bufA  = (float*)d_ws;                  // 16,777,216 floats
    float* bufB  = bufA + (size_t)16777216;       // 16,777,216 floats
    float* st    = bufB + (size_t)16777216;
    float* mean  = st;                            // up to 512
    float* rstd  = st + 1024;                     // up to 512
    float* qbuf  = st + 2048;                     // 256
    float* qscr  = st + 2304;                     // 256
    float* norms = st + 2560;                     // 256
    float* maskb = st + 2816;                     // 256

    const int B = 256;

    auto conv = [&](const float* in, const float* wt, const float* bi,
                    const float* mk, float* o, int Cin, int Cout, int H, int W) {
        int Npix = B * H * W;
        dim3 grid((Npix + 63) / 64, Cout / 32);
        conv3x3_wmma_kernel<<<grid, dim3(128), 0, stream>>>(
            in, wt, bi, mk, o, Cin, Cout, H, W, B);
    };
    auto bn2 = [&](float* a, const float* gg, const float* bbe, int C, int H, int W) {
        int HW = H * W;
        bn2d_stats_kernel<<<dim3(C), dim3(256), 0, stream>>>(a, mean, rstd, C, HW, B);
        long total = (long)B * C * HW;
        bn2d_apply_relu_kernel<<<dim3((unsigned)((total + 255) / 256)), dim3(256), 0, stream>>>(
            a, mean, rstd, gg, bbe, C, HW, total);
    };
    auto pool = [&](const float* in, float* o, int C, int H, int W) {
        long total = (long)B * C * (H / 2) * (W / 2);
        maxpool2_kernel<<<dim3((unsigned)((total + 255) / 256)), dim3(256), 0, stream>>>(
            in, o, C, H, W, B);
    };
    auto gemm = [&](const float* A, const float* Wt, const float* bi, float* o,
                    int M, int N, int K) {
        dim3 grid((N + 63) / 64, (M + 31) / 32);
        gemm_wmma_kernel<<<grid, dim3(128), 0, stream>>>(A, Wt, bi, o, M, N, K);
    };
    auto bn1 = [&](float* a, const float* gg, const float* bbe, int M, int F) {
        bn1d_stats_kernel<<<dim3(F), dim3(256), 0, stream>>>(a, mean, rstd, M, F);
        long total = (long)M * F;
        bn1d_apply_relu_kernel<<<dim3((unsigned)((total + 255) / 256)), dim3(256), 0, stream>>>(
            a, mean, rstd, gg, bbe, F, total);
    };

    // Layer 1-2 (32x32, C=64)
    conv(x,    w[1], bb[1], nullptr, bufA, 3,  64, 32, 32);
    bn2(bufA, g[1], be[1], 64, 32, 32);
    conv(bufA, w[2], bb[2], nullptr, bufB, 64, 64, 32, 32);
    bn2(bufB, g[2], be[2], 64, 32, 32);
    pool(bufB, bufA, 64, 32, 32);                 // -> (B,64,16,16)

    // Layer 3-4 (16x16, C=128)
    conv(bufA, w[3], bb[3], nullptr, bufB, 64, 128, 16, 16);
    bn2(bufB, g[3], be[3], 128, 16, 16);
    conv(bufB, w[4], bb[4], nullptr, bufA, 128, 128, 16, 16);
    bn2(bufA, g[4], be[4], 128, 16, 16);
    pool(bufA, bufB, 128, 16, 16);                // -> (B,128,8,8)

    // Layer 5 (8x8, C=256)
    conv(bufB, w[5], bb[5], nullptr, bufA, 128, 256, 8, 8);
    bn2(bufA, g[5], be[5], 256, 8, 8);            // bufA = x5 (post-relu)

    // ALSH mask for conv6
    bn2d_stats_kernel<<<dim3(256), dim3(256), 0, stream>>>(bufA, qbuf, qscr, 256, 64, B);
    filter_norms_kernel<<<dim3(256), dim3(256), 0, stream>>>(w[6], norms, 2304);
    alsh_mask_kernel<<<dim3(1), dim3(256), 0, stream>>>(w[6], hash_a, norms, qbuf,
                                                        maskb, 2304, 256);

    // Layer 6 (masked)
    conv(bufA, w[6], bb[6], maskb, bufB, 256, 256, 8, 8);
    bn2(bufB, g[6], be[6], 256, 8, 8);
    pool(bufB, bufA, 256, 8, 8);                  // -> (B,256,4,4) == (B,4096)

    // FC head
    gemm(bufA, fc7_w, fc7_b, bufB, 256, 512, 4096);
    bn1(bufB, g7, be7, 256, 512);
    gemm(bufB, fc8_w, fc8_b, bufA, 256, 512, 512);
    bn1(bufA, g8, be8, 256, 512);
    gemm(bufA, fc9_w, fc9_b, out, 256, 10, 512);
}
